// RelationAwareTransformerLayer_55078660604308
// MI455X (gfx1250) — matrix operand order repaired and checked by reading, not compiled, not run on hardware
//
#include <hip/hip_runtime.h>

// ---------------------------------------------------------------- types
typedef __bf16 bf16_t;
typedef __attribute__((ext_vector_type(16))) __bf16 v16bf;
typedef __attribute__((ext_vector_type(8)))  __bf16 v8bf;
typedef __attribute__((ext_vector_type(4)))  __bf16 v4bf;
typedef __attribute__((ext_vector_type(8)))  float  v8f;

static constexpr int Dm = 1024;   // token dim
static constexpr int Hn = 16;     // heads
static constexpr int HDm = 64;    // head dim
static constexpr int FFm = 4096;  // ff dim
static constexpr int Bn = 4;
static constexpr int Sn = 1024;
static constexpr int BSn = Bn * Sn;   // 4096 rows

// ---------------------------------------------------------------- helpers
__device__ __forceinline__ bf16_t f2bf(float f) {
  unsigned u = __float_as_uint(f);
  unsigned r = (u + 0x7FFFu + ((u >> 16) & 1u)) >> 16;
  unsigned short s = (unsigned short)r;
  return __builtin_bit_cast(bf16_t, s);
}
__device__ __forceinline__ float bf2f(bf16_t h) {
  unsigned short s = __builtin_bit_cast(unsigned short, h);
  return __uint_as_float(((unsigned)s) << 16);
}
__device__ __forceinline__ float wave_sum(float v) {
  for (int o = 16; o > 0; o >>= 1) v += __shfl_xor(v, o, 32);
  return v;
}
__device__ __forceinline__ float wave_max(float v) {
  for (int o = 16; o > 0; o >>= 1) v = fmaxf(v, __shfl_xor(v, o, 32));
  return v;
}
// CDNA5 16-bit A-fragment (16x32): lane<16 holds K {k0..k0+7, k0+16..k0+23},
// lane>=16 holds K {k0+8..k0+15, k0+24..k0+31}. Two 16B chunks per lane.
__device__ __forceinline__ v16bf ldA(const bf16_t* rowp, int k0, int hi) {
  v8bf lo = *(const v8bf*)(rowp + k0 + hi * 8);
  v8bf hh = *(const v8bf*)(rowp + k0 + 16 + hi * 8);
  return __builtin_shufflevector(lo, hh, 0,1,2,3,4,5,6,7,8,9,10,11,12,13,14,15);
}
// CDNA5 16-bit B-fragment (32x16): lane = column, same K order as A-fragment.
__device__ __forceinline__ v16bf ldB(const bf16_t* colp, int k0, int hi) {
  v8bf lo = *(const v8bf*)(colp + k0 + hi * 8);
  v8bf hh = *(const v8bf*)(colp + k0 + 16 + hi * 8);
  return __builtin_shufflevector(lo, hh, 0,1,2,3,4,5,6,7,8,9,10,11,12,13,14,15);
}
__device__ __forceinline__ v8f wmma_bf16(v16bf a, v16bf b, v8f c) {
  return __builtin_amdgcn_wmma_f32_16x16x32_bf16(false, a, false, b, (short)0, c,
                                                 false, false);
}
__device__ __forceinline__ float gelu_exact(float x) {
  return 0.5f * x * (1.0f + erff(x * 0.70710678118654752f));
}

// ---------------------------------------------------------------- fp32 -> bf16 (x4)
__global__ __launch_bounds__(256) void k_cvt_bf16(const float* __restrict__ in,
                                                  bf16_t* __restrict__ out, int n) {
  int i = (blockIdx.x * 256 + threadIdx.x) * 4;
  if (i < n) {
    float4 v = *(const float4*)(in + i);
    v4bf o;
    o[0] = f2bf(v.x); o[1] = f2bf(v.y); o[2] = f2bf(v.z); o[3] = f2bf(v.w);
    *(v4bf*)(out + i) = o;
  }
}

// ---------------------------------------------------------------- layernorm (D=1024)
__global__ __launch_bounds__(256) void k_layernorm(const float* __restrict__ x,
                                                   const float* __restrict__ g,
                                                   const float* __restrict__ b,
                                                   bf16_t* __restrict__ out) {
  __shared__ float red[8];
  const int row = blockIdx.x;
  const float4* x4 = (const float4*)(x + (size_t)row * Dm);
  float4 vv = x4[threadIdx.x];
  float v[4] = {vv.x, vv.y, vv.z, vv.w};
  const int lane = threadIdx.x & 31, wid = threadIdx.x >> 5;

  float s = v[0] + v[1] + v[2] + v[3];
  s = wave_sum(s);
  if (lane == 0) red[wid] = s;
  __syncthreads();
  float mean = 0.f;
#pragma unroll
  for (int i = 0; i < 8; i++) mean += red[i];
  mean *= (1.0f / 1024.0f);
  __syncthreads();

  float s2 = 0.f;
#pragma unroll
  for (int j = 0; j < 4; j++) { float d = v[j] - mean; s2 += d * d; }
  s2 = wave_sum(s2);
  if (lane == 0) red[wid] = s2;
  __syncthreads();
  float var = 0.f;
#pragma unroll
  for (int i = 0; i < 8; i++) var += red[i];
  var *= (1.0f / 1024.0f);
  const float rstd = rsqrtf(var + 1e-5f);

  const int c0 = threadIdx.x * 4;
  v4bf o;
#pragma unroll
  for (int j = 0; j < 4; j++) o[j] = f2bf((v[j] - mean) * rstd * g[c0 + j] + b[c0 + j]);
  *(v4bf*)(out + (size_t)row * Dm + c0) = o;
}

// ---------------------------------------------------------------- generic WMMA GEMM
// C[M,N] = A[M,K](bf16) @ W[N,K]^T(bf16) + bias[N]
// MODE 0: bf16 out; MODE 1: bf16 out with exact GELU; MODE 2: fp32 out + fp32 residual
template <int MODE>
__global__ __launch_bounds__(256) void k_gemm(const bf16_t* __restrict__ A,
                                              const bf16_t* __restrict__ W,
                                              const float* __restrict__ bias,
                                              const float* __restrict__ resid,
                                              void* __restrict__ outp,
                                              int M, int N, int K) {
  const int tid = threadIdx.x;
  const int lane = tid & 31, wid = tid >> 5;
  const int wrow = wid >> 2, wcol = wid & 3;             // 2x4 wave grid
  const int m0 = blockIdx.y * 128 + wrow * 64;
  const int n0 = blockIdx.x * 128 + wcol * 32;
  const int lm = lane & 15, hi = lane >> 4;

  v8f acc[4][2];
  const v8f z = {};
#pragma unroll
  for (int i = 0; i < 4; i++)
#pragma unroll
    for (int j = 0; j < 2; j++) acc[i][j] = z;

  const bf16_t* Arow = A + (size_t)(m0 + lm) * K;
  const bf16_t* Wrow = W + (size_t)(n0 + lm) * K;
  const size_t astep = (size_t)16 * K;

  for (int k0 = 0; k0 < K; k0 += 32) {
    v16bf a[4], b[2];
#pragma unroll
    for (int i = 0; i < 4; i++) a[i] = ldA(Arow + (size_t)i * astep, k0, hi);
#pragma unroll
    for (int j = 0; j < 2; j++) b[j] = ldB(Wrow + (size_t)j * astep, k0, hi);
#pragma unroll
    for (int i = 0; i < 4; i++)
#pragma unroll
      for (int j = 0; j < 2; j++) acc[i][j] = wmma_bf16(a[i], b[j], acc[i][j]);
  }
  // C/D layout: vgpr r, lane L -> row = r + 8*(L>>4), col = L&15
#pragma unroll
  for (int i = 0; i < 4; i++)
#pragma unroll
    for (int j = 0; j < 2; j++) {
      const int mmb = m0 + i * 16 + hi * 8;
      const int nn = n0 + j * 16 + lm;
      const float bv = bias[nn];
#pragma unroll
      for (int r = 0; r < 8; r++) {
        float v = acc[i][j][r] + bv;
        size_t idx = (size_t)(mmb + r) * N + nn;
        if (MODE == 2) {
          ((float*)outp)[idx] = v + resid[idx];
        } else {
          if (MODE == 1) v = gelu_exact(v);
          ((bf16_t*)outp)[idx] = f2bf(v);
        }
      }
    }
}

// ---------------------------------------------------------------- V transpose
// v[B,S,H*HD] -> vt[B,H,HD,S]
__global__ __launch_bounds__(256) void k_transpose_v(const bf16_t* __restrict__ v,
                                                     bf16_t* __restrict__ vt) {
  size_t i = (size_t)blockIdx.x * 256 + threadIdx.x;  // over B*H*HD*S = 4M
  int s = (int)(i & 1023);
  size_t j = i >> 10;
  int d = (int)(j & 63); j >>= 6;
  int h = (int)(j & 15);
  int b = (int)(j >> 4);
  vt[i] = v[(((size_t)(b * Sn + s)) << 10) + h * HDm + d];
}

// ---------------------------------------------------------------- fused attention
// grid: (S/64, B*H). Per block: 64 query rows, full t-range in LDS (133 KB).
__global__ __launch_bounds__(256) void k_attn(
    const bf16_t* __restrict__ q, const bf16_t* __restrict__ kk,
    const bf16_t* __restrict__ vt,
    const float* __restrict__ attn_bias,  // [B,S,S,3]
    const float* __restrict__ meta,       // [B,S,S]
    const float* __restrict__ lat,        // [B,H,S,S]
    const float* __restrict__ mrb,        // [B,H,S,S]
    const float* __restrict__ Wsb, const float* __restrict__ bsb,
    const float* __restrict__ Wmb, const float* __restrict__ bmb,
    const unsigned char* __restrict__ mask,  // [B,S]
    bf16_t* __restrict__ attended)           // [B,S,D]
{
  constexpr int RS = 1040;                 // padded row stride (bf16 elems)
  __shared__ bf16_t sm[64 * RS];           // 133,120 bytes (MI455X: 320KB/WGP)

  const int tid = threadIdx.x, lane = tid & 31, wid = tid >> 5;
  const int lm = lane & 15, hi = lane >> 4;
  const int bh = blockIdx.y, b = bh >> 4, h = bh & 15;
  const int s0 = blockIdx.x * 64;
  const float scale = 0.125f;  // 64^-0.5
  const float wsb0 = Wsb[h * 3], wsb1 = Wsb[h * 3 + 1], wsb2 = Wsb[h * 3 + 2];
  const float sb = bsb[h], wmbf = Wmb[h], mbf = bmb[h];
  const size_t bhSS = (size_t)bh << 20;

  // ---- phase 1: logits = scale*QK^T + biases -> bf16 in LDS
  {
    const int rstrip = wid >> 1;       // 4 strips of 16 rows
    const int thalf = wid & 1;         // each wave does 512 of t
    const int srow = s0 + rstrip * 16;
    const bf16_t* qrow = q + (((size_t)(b * Sn + srow + lm)) << 10) + h * HDm;
    v16bf a0 = ldA(qrow, 0, hi);
    v16bf a1 = ldA(qrow, 32, hi);
    for (int t0 = thalf * 512; t0 < thalf * 512 + 512; t0 += 16) {
      const bf16_t* krow = kk + (((size_t)(b * Sn + t0 + lm)) << 10) + h * HDm;
      v8f acc = {};
      acc = wmma_bf16(a0, ldB(krow, 0, hi), acc);
      acc = wmma_bf16(a1, ldB(krow, 32, hi), acc);
      const int tt = t0 + lm;
      const bool msk = mask[b * Sn + tt] != 0;
      const int slb = rstrip * 16 + hi * 8;
      const size_t stb = (((size_t)(b * Sn + s0 + slb)) << 10) + tt;
      const size_t btb = bhSS + (((size_t)(s0 + slb)) << 10) + tt;
#pragma unroll
      for (int r = 0; r < 8; r++) {
        const size_t st = stb + ((size_t)r << 10);
        const size_t bt = btb + ((size_t)r << 10);
        float lg = acc[r] * scale;
        const float* ab = attn_bias + st * 3;
        lg += ab[0] * wsb0 + ab[1] * wsb1 + ab[2] * wsb2 + sb;
        lg += meta[st] * wmbf + mbf;
        lg += lat[bt] + mrb[bt];
        if (msk) lg = -1e9f;
        sm[(slb + r) * RS + tt] = f2bf(lg);
      }
    }
  }
  __syncthreads();

  // ---- phase 2: softmax in LDS, vectorized 16B chunks (8 rows per wave)
  for (int rr = 0; rr < 8; rr++) {
    v8bf* rowv = (v8bf*)(sm + (wid * 8 + rr) * RS);   // 128 chunks of 8 used
    float mx = -3.0e38f;
#pragma unroll
    for (int c = 0; c < 4; c++) {
      v8bf ch = rowv[lane + 32 * c];
#pragma unroll
      for (int e = 0; e < 8; e++) mx = fmaxf(mx, bf2f(ch[e]));
    }
    mx = wave_max(mx);
    float sum = 0.f;
#pragma unroll
    for (int c = 0; c < 4; c++) {
      v8bf ch = rowv[lane + 32 * c];
      v8bf eo;
#pragma unroll
      for (int e = 0; e < 8; e++) {
        float ex = __expf(bf2f(ch[e]) - mx);
        sum += ex;
        eo[e] = f2bf(ex);
      }
      rowv[lane + 32 * c] = eo;
    }
    sum = wave_sum(sum);
    const float inv = 1.0f / sum;
#pragma unroll
    for (int c = 0; c < 4; c++) {
      v8bf ch = rowv[lane + 32 * c];
#pragma unroll
      for (int e = 0; e < 8; e++) ch[e] = f2bf(bf2f(ch[e]) * inv);
      rowv[lane + 32 * c] = ch;
    }
  }
  __syncthreads();

  // ---- phase 3: attended = P @ V  (A-fragments from LDS, B from V^T)
  {
    const int rstrip = wid >> 1;
    const int chalf = wid & 1;  // 32 of the 64 head-dim cols
    v8f acc[2];
    const v8f z = {};
    acc[0] = z; acc[1] = z;
    const bf16_t* arow = sm + (rstrip * 16 + lm) * RS;
    const bf16_t* vrow = vt + (((size_t)bh * HDm + chalf * 32 + lm) << 10);
    for (int k0 = 0; k0 < Sn; k0 += 32) {
      v16bf a = ldA(arow, k0, hi);
      acc[0] = wmma_bf16(a, ldB(vrow, k0, hi), acc[0]);
      acc[1] = wmma_bf16(a, ldB(vrow + ((size_t)16 << 10), k0, hi), acc[1]);
    }
#pragma unroll
    for (int j = 0; j < 2; j++) {
      const int col = h * HDm + chalf * 32 + j * 16 + lm;
#pragma unroll
      for (int r = 0; r < 8; r++) {
        const int sg = s0 + rstrip * 16 + hi * 8 + r;
        attended[(((size_t)(b * Sn + sg)) << 10) + col] = f2bf(acc[j][r]);
      }
    }
  }
}

// ---------------------------------------------------------------- launch
extern "C" void kernel_launch(void* const* d_in, const int* in_sizes, int n_in,
                              void* d_out, int out_size, void* d_ws, size_t ws_size,
                              hipStream_t stream) {
  const float* tokens = (const float*)d_in[0];
  const unsigned char* mask = (const unsigned char*)d_in[1];
  const float* attn_bias = (const float*)d_in[2];
  const float* meta = (const float*)d_in[3];
  const float* lat = (const float*)d_in[4];
  const float* mrb = (const float*)d_in[5];
  const float* ln1_g = (const float*)d_in[6];
  const float* ln1_b = (const float*)d_in[7];
  const float* Wq = (const float*)d_in[8];  const float* bq = (const float*)d_in[9];
  const float* Wk = (const float*)d_in[10]; const float* bk = (const float*)d_in[11];
  const float* Wv = (const float*)d_in[12]; const float* bv = (const float*)d_in[13];
  const float* Wo = (const float*)d_in[14]; const float* bo = (const float*)d_in[15];
  const float* Wsb = (const float*)d_in[16]; const float* bsb = (const float*)d_in[17];
  const float* Wmb = (const float*)d_in[18]; const float* bmb = (const float*)d_in[19];
  const float* ln2_g = (const float*)d_in[20]; const float* ln2_b = (const float*)d_in[21];
  const float* W1 = (const float*)d_in[22]; const float* b1 = (const float*)d_in[23];
  const float* W2 = (const float*)d_in[24]; const float* b2 = (const float*)d_in[25];
  float* out = (float*)d_out;

  // -------- workspace carve-up (~128 MB)
  char* p = (char*)d_ws;
  auto take = [&](size_t bytes) { char* r = p; p += (bytes + 255) & ~(size_t)255; return r; };
  const size_t MB2 = (size_t)1048576 * 2;
  bf16_t* wq_b = (bf16_t*)take(MB2);           // 1M bf16 each
  bf16_t* wk_b = (bf16_t*)take(MB2);
  bf16_t* wv_b = (bf16_t*)take(MB2);
  bf16_t* wo_b = (bf16_t*)take(MB2);
  bf16_t* w1_b = (bf16_t*)take(4 * MB2);       // 4M
  bf16_t* w2_b = (bf16_t*)take(4 * MB2);       // 4M
  bf16_t* xn   = (bf16_t*)take(4 * MB2);       // normalized tokens
  bf16_t* qb   = (bf16_t*)take(4 * MB2);
  bf16_t* kb   = (bf16_t*)take(4 * MB2);
  bf16_t* vb   = (bf16_t*)take(4 * MB2);
  bf16_t* vtb  = (bf16_t*)take(4 * MB2);       // V^T [B,H,HD,S]
  bf16_t* att  = (bf16_t*)take(4 * MB2);       // attended bf16
  float*  tok2 = (float*)take((size_t)BSn * Dm * 4);  // post-attn residual (fp32)
  bf16_t* h2   = (bf16_t*)take(4 * MB2);       // ln2 output
  bf16_t* ffm  = (bf16_t*)take(16 * MB2);      // gelu(h@W1^T+b1), [4096,4096]

  // -------- weight conversions (vectorized x4)
  k_cvt_bf16<<<(Dm * Dm) / 1024, 256, 0, stream>>>(Wq, wq_b, Dm * Dm);
  k_cvt_bf16<<<(Dm * Dm) / 1024, 256, 0, stream>>>(Wk, wk_b, Dm * Dm);
  k_cvt_bf16<<<(Dm * Dm) / 1024, 256, 0, stream>>>(Wv, wv_b, Dm * Dm);
  k_cvt_bf16<<<(Dm * Dm) / 1024, 256, 0, stream>>>(Wo, wo_b, Dm * Dm);
  k_cvt_bf16<<<(FFm * Dm) / 1024, 256, 0, stream>>>(W1, w1_b, FFm * Dm);
  k_cvt_bf16<<<(Dm * FFm) / 1024, 256, 0, stream>>>(W2, w2_b, Dm * FFm);

  // -------- LN1
  k_layernorm<<<BSn, 256, 0, stream>>>(tokens, ln1_g, ln1_b, xn);

  // -------- Q, K, V projections (WMMA)
  dim3 g_proj(Dm / 128, BSn / 128);
  k_gemm<0><<<g_proj, 256, 0, stream>>>(xn, wq_b, bq, nullptr, qb, BSn, Dm, Dm);
  k_gemm<0><<<g_proj, 256, 0, stream>>>(xn, wk_b, bk, nullptr, kb, BSn, Dm, Dm);
  k_gemm<0><<<g_proj, 256, 0, stream>>>(xn, wv_b, bv, nullptr, vb, BSn, Dm, Dm);
  k_transpose_v<<<(Bn * Hn * HDm * Sn) / 256, 256, 0, stream>>>(vb, vtb);

  // -------- fused attention (logits + biases + softmax + PV in LDS)
  dim3 g_attn(Sn / 64, Bn * Hn);
  k_attn<<<g_attn, 256, 0, stream>>>(qb, kb, vtb, attn_bias, meta, lat, mrb,
                                     Wsb, bsb, Wmb, bmb, mask, att);

  // -------- Wo projection + residual (fp32 out)
  k_gemm<2><<<g_proj, 256, 0, stream>>>(att, wo_b, bo, tokens, tok2, BSn, Dm, Dm);

  // -------- LN2 + FFN
  k_layernorm<<<BSn, 256, 0, stream>>>(tok2, ln2_g, ln2_b, h2);
  dim3 g_ff1(FFm / 128, BSn / 128);
  k_gemm<1><<<g_ff1, 256, 0, stream>>>(h2, w1_b, b1, nullptr, ffm, BSn, FFm, Dm);
  dim3 g_ff2(Dm / 128, BSn / 128);
  k_gemm<2><<<g_ff2, 256, 0, stream>>>(ffm, w2_b, b2, tok2, out, BSn, Dm, FFm);
}
